// GraphNN_71055938945206
// MI455X (gfx1250) — matrix operand order repaired and checked by reading, not compiled, not run on hardware
//
#include <hip/hip_runtime.h>
#include <hip/hip_bf16.h>

#define NNODES 50000
#define NGRAPH 100
#define NPG    500
#define NE     400000
#define NP     200000

typedef _Float16 v16h  __attribute__((ext_vector_type(16)));
typedef _Float16 half8 __attribute__((ext_vector_type(8)));
typedef float    v8f   __attribute__((ext_vector_type(8)));

__device__ __forceinline__ unsigned f2key(float f) {
  unsigned u = __float_as_uint(f);
  return (u & 0x80000000u) ? ~u : (u | 0x80000000u);
}
__device__ __forceinline__ float key2f(unsigned k) {
  unsigned u = (k & 0x80000000u) ? (k ^ 0x80000000u) : ~k;
  return __uint_as_float(u);
}

// Async DMA one 16-byte chunk global -> LDS (ASYNCcnt-tracked, gfx1250).
__device__ __forceinline__ void async_g2l_b128(void* lds, const void* gmem) {
  unsigned l = (unsigned)(uintptr_t)lds;  // low 32 bits of generic ptr = LDS byte address
  asm volatile("global_load_async_to_lds_b128 %0, %1, off"
               :: "v"(l), "v"(gmem) : "memory");
}
__device__ __forceinline__ void wait_async0() {
  asm volatile("s_wait_asynccnt 0x0" ::: "memory");
}

__global__ void k_zero(float* __restrict__ p, int n) {
  int i = blockIdx.x * blockDim.x + threadIdx.x;
  if (i < n) p[i] = 0.f;
}

// Convert Wd0 (257x128) -> w0t[128][288] f16 with permuted K = [src128 | dst128 | w | pad]
// Convert Wd1 (128x64)  -> w1t[64][128] f16 (column-major: w1t[n][k] = Wd1[k][n])
__global__ void k_prep_w(const float* __restrict__ Wd0, const float* __restrict__ Wd1,
                         _Float16* __restrict__ w0t, _Float16* __restrict__ w1t) {
  int i = blockIdx.x * blockDim.x + threadIdx.x;
  if (i < 128 * 288) {
    int n = i / 288, k = i % 288;
    float v = 0.f;
    if (k < 128)       v = Wd0[k * 128 + n];          // x[src] rows 0..127
    else if (k < 256)  v = Wd0[(k + 1) * 128 + n];    // x[dst] rows 129..256
    else if (k == 256) v = Wd0[128 * 128 + n];        // w row 128
    w0t[n * 288 + k] = (_Float16)v;
  } else if (i < 128 * 288 + 64 * 128) {
    int j = i - 128 * 288;
    int n = j / 128, k = j % 128;
    w1t[n * 128 + k] = (_Float16)Wd1[k * 64 + n];
  }
}

// GNN scatter: agg[dst][c] += x[src][c] * w[e]
__global__ void k_scatter(const float* __restrict__ x, const int* __restrict__ edges,
                          const float* __restrict__ ea, float* __restrict__ agg, int D) {
  long i = (long)blockIdx.x * blockDim.x + threadIdx.x;
  if (i >= (long)NE * D) return;
  int e = (int)(i / D), c = (int)(i % D);
  int s = edges[e], d = edges[NE + e];
  atomicAdd(&agg[(long)d * D + c], x[(long)s * D + c] * ea[(long)e * 2]);
}

// x_new = tanh(agg @ Wrel + brel + x @ Wroot)   (small dims -> VALU)
__global__ void k_node(const float* __restrict__ x, const float* __restrict__ agg,
                       const float* __restrict__ Wr, const float* __restrict__ br,
                       const float* __restrict__ Wrt, float* __restrict__ outf,
                       _Float16* __restrict__ outh, int Din, int Dout) {
  long i = (long)blockIdx.x * blockDim.x + threadIdx.x;
  if (i >= (long)NNODES * Dout) return;
  int n = (int)(i / Dout), j = (int)(i % Dout);
  float acc = br[j];
  for (int k = 0; k < Din; ++k)
    acc += agg[(long)n * Din + k] * Wr[k * Dout + j] +
           x[(long)n * Din + k] * Wrt[k * Dout + j];
  float t = tanhf(acc);
  if (outf) outf[i] = t;
  if (outh) outh[i] = (_Float16)t;
}

// One aligned 8-half run of the permuted-K feature row [src(0..127) | dst(128..255) | w(256) | 0]
__device__ __forceinline__ half8 load_run0(const _Float16* xs, const _Float16* xd,
                                           float w, int ks) {
  if (ks + 8 <= 128) return *(const half8*)(xs + ks);
  if (ks + 8 <= 256) return *(const half8*)(xd + (ks - 128));
  half8 r;
#pragma unroll
  for (int j = 0; j < 8; ++j) {
    int k = ks + j;
    r[j] = (k == 256) ? (_Float16)w : (_Float16)0.f;
  }
  return r;
}

// Edge GEMM0: f1 = tanh(feat(257) @ Wd0 + bd0), fused per-(graph,channel) sum/sumsq atomics.
// One wave = 16 edges x all 128 output channels; 9 K-chunks of 32.
// Full 72 KB B panel staged in LDS (320 KB/WGP budget -> 4 blocks/WGP).
__global__ __launch_bounds__(256) void k_edge_gemm0(
    const _Float16* __restrict__ h3, const int* __restrict__ edges,
    const float* __restrict__ ea, const _Float16* __restrict__ w0t,
    const float* __restrict__ bd0, _Float16* __restrict__ f1,
    float* __restrict__ ssum, float* __restrict__ ssq) {
  __shared__ __align__(32) _Float16 bsh0[128 * 288];   // 73728 B
  for (int t = threadIdx.x; t < 128 * 288 / 8; t += 256)
    ((half8*)bsh0)[t] = ((const half8*)w0t)[t];
  __syncthreads();

  int lane = threadIdx.x & 31, wid = threadIdx.x >> 5;
  int tile = blockIdx.x * 8 + wid;
  if (tile >= NE / 16) return;
  int m0 = tile * 16, hi = lane >> 4, l15 = lane & 15;
  int row = m0 + l15;
  int s = edges[row], d = edges[NE + row];
  const _Float16* xs = h3 + (long)s * 128;
  const _Float16* xd = h3 + (long)d * 128;
  float w = ea[(long)row * 2];

  v8f c[8] = {};
#pragma unroll
  for (int kc = 0; kc < 9; ++kc) {
    int k0 = kc * 32;
    int ks0 = k0 + hi * 8;
    half8 r0 = load_run0(xs, xd, w, ks0);
    half8 r1 = load_run0(xs, xd, w, ks0 + 16);
    v16h a;
#pragma unroll
    for (int j = 0; j < 8; ++j) { a[j] = r0[j]; a[j + 8] = r1[j]; }
#pragma unroll
    for (int nt = 0; nt < 8; ++nt) {
      const _Float16* bp = bsh0 + (nt * 16 + l15) * 288 + k0 + hi * 8;
      half8 b0 = *(const half8*)bp, b1 = *(const half8*)(bp + 16);
      v16h b;
#pragma unroll
      for (int j = 0; j < 8; ++j) { b[j] = b0[j]; b[j + 8] = b1[j]; }
      c[nt] = __builtin_amdgcn_wmma_f32_16x16x32_f16(false, a, false, b,
                                                     (short)0, c[nt], false, false);
    }
  }
  int g = edges[m0] / NPG;  // 4000 edges/graph, 16-edge tiles never cross graphs
#pragma unroll
  for (int nt = 0; nt < 8; ++nt) {
    int chan = nt * 16 + l15;
    float bias = bd0[chan];
    float s1 = 0.f, s2 = 0.f;
#pragma unroll
    for (int v = 0; v < 8; ++v) {
      float t = tanhf(c[nt][v] + bias);
      f1[(long)(m0 + v + hi * 8) * 128 + chan] = (_Float16)t;
      s1 += t; s2 += t * t;
    }
    s1 += __shfl_xor(s1, 16, 32);
    s2 += __shfl_xor(s2, 16, 32);
    if (hi == 0) {
      atomicAdd(&ssum[g * 128 + chan], s1);
      atomicAdd(&ssq[g * 128 + chan], s2);
    }
  }
}

// stats -> per-(graph,channel) affine coefs: out = cA*h + cB
__global__ void k_coef(const float* __restrict__ ssum, const float* __restrict__ ssq,
                       const float* __restrict__ gnw, const float* __restrict__ gnb,
                       const float* __restrict__ gnms, float cnt, int C,
                       float* __restrict__ cA, float* __restrict__ cB) {
  int i = blockIdx.x * blockDim.x + threadIdx.x;
  if (i >= NGRAPH * C) return;
  int ch = i % C;
  float mean = ssum[i] / cnt;
  float eh2  = ssq[i] / cnt;
  float ms   = gnms[ch];
  float var  = eh2 - 2.f * ms * mean * mean + ms * ms * mean * mean;
  float rsq  = rsqrtf(var + 1e-5f);
  cA[i] = gnw[ch] * rsq;
  cB[i] = gnb[ch] - gnw[ch] * rsq * mean * ms;
}

__global__ void k_norm0(const _Float16* __restrict__ f1, const int* __restrict__ edges,
                        const float* __restrict__ cA, const float* __restrict__ cB,
                        _Float16* __restrict__ a1) {
  long i = (long)blockIdx.x * blockDim.x + threadIdx.x;
  if (i >= (long)NE * 128) return;
  int e = (int)(i >> 7), ch = (int)(i & 127);
  int g = edges[e] / NPG;
  a1[i] = (_Float16)(cA[g * 128 + ch] * (float)f1[i] + cB[g * 128 + ch]);
}

// Edge GEMM1: f2 = tanh(a1(128) @ Wd1 + bd1); B panel DMA'd into LDS with
// gfx1250 async global->LDS copies; fused stats.
__global__ __launch_bounds__(256) void k_edge_gemm1(
    const _Float16* __restrict__ a1, const int* __restrict__ edges,
    const _Float16* __restrict__ w1t, const float* __restrict__ bd1,
    _Float16* __restrict__ f2, float* __restrict__ ssum, float* __restrict__ ssq) {
  __shared__ __align__(32) _Float16 bsh[64 * 128];     // 16 KB
  for (int t = threadIdx.x; t < 64 * 128 / 8; t += 256)
    async_g2l_b128(bsh + t * 8, w1t + t * 8);
  wait_async0();
  __syncthreads();

  int lane = threadIdx.x & 31, wid = threadIdx.x >> 5;
  int tile = blockIdx.x * 8 + wid;
  if (tile >= NE / 16) return;
  int m0 = tile * 16, hi = lane >> 4, l15 = lane & 15;
  int row = m0 + l15;
  const _Float16* ar = a1 + (long)row * 128;

  v8f c[4] = {};
#pragma unroll
  for (int kc = 0; kc < 4; ++kc) {
    int k0 = kc * 32, ks0 = k0 + hi * 8;
    half8 r0 = *(const half8*)(ar + ks0);
    half8 r1 = *(const half8*)(ar + ks0 + 16);
    v16h a;
#pragma unroll
    for (int j = 0; j < 8; ++j) { a[j] = r0[j]; a[j + 8] = r1[j]; }
#pragma unroll
    for (int nt = 0; nt < 4; ++nt) {
      const _Float16* bp = bsh + (nt * 16 + l15) * 128 + k0 + hi * 8;
      half8 b0 = *(const half8*)bp, b1 = *(const half8*)(bp + 16);
      v16h b;
#pragma unroll
      for (int j = 0; j < 8; ++j) { b[j] = b0[j]; b[j + 8] = b1[j]; }
      c[nt] = __builtin_amdgcn_wmma_f32_16x16x32_f16(false, a, false, b,
                                                     (short)0, c[nt], false, false);
    }
  }
  int g = edges[m0] / NPG;
#pragma unroll
  for (int nt = 0; nt < 4; ++nt) {
    int chan = nt * 16 + l15;
    float bias = bd1[chan];
    float s1 = 0.f, s2 = 0.f;
#pragma unroll
    for (int v = 0; v < 8; ++v) {
      float t = tanhf(c[nt][v] + bias);
      f2[(long)(m0 + v + hi * 8) * 64 + chan] = (_Float16)t;
      s1 += t; s2 += t * t;
    }
    s1 += __shfl_xor(s1, 16, 32);
    s2 += __shfl_xor(s2, 16, 32);
    if (hi == 0) {
      atomicAdd(&ssum[g * 64 + chan], s1);
      atomicAdd(&ssq[g * 64 + chan], s2);
    }
  }
}

// vals[e] = bout + sum_c Wout[c] * gnorm1(f2[e][c])
__global__ void k_final(const _Float16* __restrict__ f2, const int* __restrict__ edges,
                        const float* __restrict__ cA, const float* __restrict__ cB,
                        const float* __restrict__ Wout, const float* __restrict__ bout,
                        float* __restrict__ vals) {
  int e = blockIdx.x * blockDim.x + threadIdx.x;
  if (e >= NE) return;
  int g = edges[e] / NPG;
  float acc = bout[0];
  for (int ch = 0; ch < 64; ++ch)
    acc += Wout[ch] * (cA[g * 64 + ch] * (float)f2[(long)e * 64 + ch] + cB[g * 64 + ch]);
  vals[e] = acc;
}

// pairwise min/argmin, emit out_edges + classes, per-graph running max (order-preserving uint key)
__global__ void k_pair(const float* __restrict__ vals, const int* __restrict__ edges,
                       const float* __restrict__ ea, float* __restrict__ out,
                       float* __restrict__ pv, unsigned* __restrict__ gmax) {
  int i = blockIdx.x * blockDim.x + threadIdx.x;
  if (i >= NP) return;
  float v0 = vals[2 * i], v1 = vals[2 * i + 1];
  int sel = (v1 < v0) ? 1 : 0;                 // argmin, ties -> 0
  float mv = sel ? v1 : v0;
  int s = edges[2 * i], d = edges[NE + 2 * i];
  out[i]          = (float)s;                  // out_edges row 0
  out[NP + i]     = (float)d;                  // out_edges row 1
  out[3 * NP + i] = ea[(long)(2 * i + sel) * 2 + 1];  // edge class
  pv[i] = mv;
  atomicMax(&gmax[s / NPG], f2key(mv));
}

__global__ void k_expsum(float* __restrict__ pv, const int* __restrict__ edges,
                         const unsigned* __restrict__ gmax, float* __restrict__ gden) {
  int i = blockIdx.x * blockDim.x + threadIdx.x;
  if (i >= NP) return;
  int g = edges[2 * i] / NPG;
  float z = expf(pv[i] - key2f(gmax[g]));
  pv[i] = z;
  atomicAdd(&gden[g], z);
}

__global__ void k_write(const float* __restrict__ pv, const int* __restrict__ edges,
                        const float* __restrict__ gden, float* __restrict__ out) {
  int i = blockIdx.x * blockDim.x + threadIdx.x;
  if (i >= NP) return;
  int g = edges[2 * i] / NPG;
  out[2 * NP + i] = pv[i] / (gden[g] + 1e-16f);
}

extern "C" void kernel_launch(void* const* d_in, const int* in_sizes, int n_in,
                              void* d_out, int out_size, void* d_ws, size_t ws_size,
                              hipStream_t stream) {
  (void)in_sizes; (void)n_in; (void)out_size; (void)ws_size;
  const float* x0     = (const float*)d_in[0];
  const int*   edges  = (const int*)d_in[1];
  const float* ea     = (const float*)d_in[2];
  const float* Wrel0  = (const float*)d_in[6];
  const float* brel0  = (const float*)d_in[7];
  const float* Wroot0 = (const float*)d_in[8];
  const float* Wrel1  = (const float*)d_in[9];
  const float* brel1  = (const float*)d_in[10];
  const float* Wroot1 = (const float*)d_in[11];
  const float* Wrel2  = (const float*)d_in[12];
  const float* brel2  = (const float*)d_in[13];
  const float* Wroot2 = (const float*)d_in[14];
  const float* Wd0    = (const float*)d_in[15];
  const float* bd0    = (const float*)d_in[16];
  const float* gnw0   = (const float*)d_in[17];
  const float* gnb0   = (const float*)d_in[18];
  const float* gnms0  = (const float*)d_in[19];
  const float* Wd1    = (const float*)d_in[20];
  const float* bd1    = (const float*)d_in[21];
  const float* gnw1   = (const float*)d_in[22];
  const float* gnb1   = (const float*)d_in[23];
  const float* gnms1  = (const float*)d_in[24];
  const float* Wout   = (const float*)d_in[25];
  const float* bout   = (const float*)d_in[26];
  float* out = (float*)d_out;

  char* p = (char*)d_ws;
  auto alloc = [&](size_t b) { void* r = (void*)p; p += (b + 255) & ~(size_t)255; return r; };
  float*    agg  = (float*)alloc((size_t)NNODES * 64 * 4);
  float*    h1   = (float*)alloc((size_t)NNODES * 32 * 4);
  float*    h2   = (float*)alloc((size_t)NNODES * 64 * 4);
  _Float16* h3   = (_Float16*)alloc((size_t)NNODES * 128 * 2);
  _Float16* w0t  = (_Float16*)alloc((size_t)128 * 288 * 2);
  _Float16* w1t  = (_Float16*)alloc((size_t)64 * 128 * 2);
  _Float16* f1   = (_Float16*)alloc((size_t)NE * 128 * 2);
  _Float16* a1   = (_Float16*)alloc((size_t)NE * 128 * 2);
  _Float16* f2   = (_Float16*)alloc((size_t)NE * 64 * 2);
  float*    st0  = (float*)alloc((size_t)2 * NGRAPH * 128 * 4);
  float*    st1  = (float*)alloc((size_t)2 * NGRAPH * 64 * 4);
  float*    cA0  = (float*)alloc((size_t)NGRAPH * 128 * 4);
  float*    cB0  = (float*)alloc((size_t)NGRAPH * 128 * 4);
  float*    cA1  = (float*)alloc((size_t)NGRAPH * 64 * 4);
  float*    cB1  = (float*)alloc((size_t)NGRAPH * 64 * 4);
  float*    vals = (float*)alloc((size_t)NE * 4);
  float*    pv   = (float*)alloc((size_t)NP * 4);
  float*    gbuf = (float*)alloc((size_t)2 * NGRAPH * 4);
  unsigned* gmax = (unsigned*)gbuf;
  float*    gden = gbuf + NGRAPH;

  const int B = 256;
  auto G = [](long n) { return (int)((n + 255) / 256); };

  k_prep_w<<<G(128 * 288 + 64 * 128), B, 0, stream>>>(Wd0, Wd1, w0t, w1t);

  // GNN layer 0: 5 -> 32
  k_zero<<<G((long)NNODES * 5), B, 0, stream>>>(agg, NNODES * 5);
  k_scatter<<<G((long)NE * 5), B, 0, stream>>>(x0, edges, ea, agg, 5);
  k_node<<<G((long)NNODES * 32), B, 0, stream>>>(x0, agg, Wrel0, brel0, Wroot0, h1, nullptr, 5, 32);
  // GNN layer 1: 32 -> 64
  k_zero<<<G((long)NNODES * 32), B, 0, stream>>>(agg, NNODES * 32);
  k_scatter<<<G((long)NE * 32), B, 0, stream>>>(h1, edges, ea, agg, 32);
  k_node<<<G((long)NNODES * 64), B, 0, stream>>>(h1, agg, Wrel1, brel1, Wroot1, h2, nullptr, 32, 64);
  // GNN layer 2: 64 -> 128 (f16 output for WMMA A gathers)
  k_zero<<<G((long)NNODES * 64), B, 0, stream>>>(agg, NNODES * 64);
  k_scatter<<<G((long)NE * 64), B, 0, stream>>>(h2, edges, ea, agg, 64);
  k_node<<<G((long)NNODES * 128), B, 0, stream>>>(h2, agg, Wrel2, brel2, Wroot2, nullptr, h3, 64, 128);

  // Edge MLP layer 0 (257 -> 128) via WMMA, LDS-staged B panel, fused gather + tanh + stats
  k_zero<<<G(2 * NGRAPH * 128), B, 0, stream>>>(st0, 2 * NGRAPH * 128);
  k_edge_gemm0<<<NE / 16 / 8, B, 0, stream>>>(h3, edges, ea, w0t, bd0, f1,
                                              st0, st0 + NGRAPH * 128);
  k_coef<<<G(NGRAPH * 128), B, 0, stream>>>(st0, st0 + NGRAPH * 128, gnw0, gnb0, gnms0,
                                            4000.f, 128, cA0, cB0);
  k_norm0<<<G((long)NE * 128), B, 0, stream>>>(f1, edges, cA0, cB0, a1);

  // Edge MLP layer 1 (128 -> 64) via WMMA with async-DMA'd LDS B panel
  k_zero<<<G(2 * NGRAPH * 64), B, 0, stream>>>(st1, 2 * NGRAPH * 64);
  k_edge_gemm1<<<NE / 16 / 8, B, 0, stream>>>(a1, edges, w1t, bd1, f2,
                                              st1, st1 + NGRAPH * 64);
  k_coef<<<G(NGRAPH * 64), B, 0, stream>>>(st1, st1 + NGRAPH * 64, gnw1, gnb1, gnms1,
                                           4000.f, 64, cA1, cB1);
  k_final<<<G(NE), B, 0, stream>>>(f2, edges, cA1, cB1, Wout, bout, vals);

  // pairing + per-graph softmax
  k_zero<<<1, B, 0, stream>>>(gbuf, 2 * NGRAPH);
  k_pair<<<G(NP), B, 0, stream>>>(vals, edges, ea, out, pv, gmax);
  k_expsum<<<G(NP), B, 0, stream>>>(pv, edges, gmax, gden);
  k_write<<<G(NP), B, 0, stream>>>(pv, edges, gden, out);
}
